// CLVBase_75282186764626
// MI455X (gfx1250) — matrix operand order repaired and checked by reading, not compiled, not run on hardware
//
#include <hip/hip_runtime.h>
#include <hip/hip_bf16.h>
#include <math.h>

typedef __attribute__((ext_vector_type(16))) __bf16 v16bf;
typedef __attribute__((ext_vector_type(8)))  float  v8f;

constexpr int Hdim = 256;   // hidden
constexpr int Edim = 8;     // experts

// Weight-fragment workspace (bf16), stream-contiguous per wave:
//   fuse : frag = ((q*16 + kt)*4 + m)*32 + lane   (q = mt/4, m = mt%4)
//   Wa   : frag = ((q*8  + kt)*4 + m)*32 + lane
//   Wg   : frag = (kt*32) + lane
constexpr int WF_FRAGS = 4 * 16 * 4 * 32;  // 256 frags * 32 lanes
constexpr int WA_FRAGS = 4 * 8 * 4 * 32;   // 128 frags * 32 lanes
constexpr int WG_FRAGS = 8 * 32;
constexpr int WF_BASE = 0;
constexpr int WA_BASE = WF_FRAGS;
constexpr int WG_BASE = WA_BASE + WA_FRAGS;
constexpr int TOTAL_ELEMS = (WF_FRAGS + WA_FRAGS + WG_FRAGS) * 16;

// LDS (dynamic) per block of 128 rows, K-split staging:
//   Xbuf: 128 rows x 256 bf16  (s_e half, then p_e half), stride 528
//   e   : 128 rows x 256 bf16, stride 528
//   g   : overlays Xbuf after fuse
constexpr int ROWS_PB   = 128;
constexpr int SSTRIDE   = 256 * 2 + 16;           // 528 (132 DW % 64 = 4 -> conflict-free)
constexpr int BUFBYTES  = ROWS_PB * SSTRIDE;      // 67584
constexpr int SMEM_BYTES = 2 * BUFBYTES;          // 135168 (2 blocks/WGP fit in 320KB)

static __device__ __forceinline__ unsigned pack_bf16(float a, float b) {
    __bf16 x = (__bf16)a, y = (__bf16)b;
    unsigned short ux = __builtin_bit_cast(unsigned short, x);
    unsigned short uy = __builtin_bit_cast(unsigned short, y);
    return (unsigned)ux | ((unsigned)uy << 16);
}
static __device__ __forceinline__ float bf_lo(unsigned u) {
    return __builtin_bit_cast(float, u << 16);
}
static __device__ __forceinline__ float bf_hi(unsigned u) {
    return __builtin_bit_cast(float, u & 0xffff0000u);
}
static __device__ __forceinline__ v16bf load_frag16(const char* p) {
    struct P { uint4 a, b; } t;
    t.a = *(const uint4*)p;
    t.b = *(const uint4*)(p + 16);
    return __builtin_bit_cast(v16bf, t);
}
static __device__ __forceinline__ float fast_tanh(float x) {
    float t = __expf(2.f * x);          // inf-safe: 1 - 2/(t+1)
    return 1.f - 2.f / (t + 1.f);
}

// ---------------------------------------------------------------------------
// Preamble: f32 weights -> bf16 WMMA A-fragments (transposed weights).
// A mapping (16-bit): lane L: m=L&15, half=L>>4;
//   e in [0,16): k = 16*half+e (contiguous, same as the B loads)
// ---------------------------------------------------------------------------
__global__ void prep_weights_kernel(const float* __restrict__ Wf,
                                    const float* __restrict__ Wa,
                                    const float* __restrict__ Wsx,
                                    const float* __restrict__ Wsg,
                                    __bf16* __restrict__ ws)
{
    int t = blockIdx.x * blockDim.x + threadIdx.x;
    if (t >= TOTAL_ELEMS) return;
    const int NF = WF_FRAGS * 16;
    const int NA = WA_FRAGS * 16;

    int idx, region;
    if (t < NF)           { region = 0; idx = t; }
    else if (t < NF + NA) { region = 1; idx = t - NF; }
    else                  { region = 2; idx = t - NF - NA; }

    int e  = idx & 15;
    int L  = (idx >> 4) & 31;
    int fi = idx >> 9;
    int halfL = L >> 4, ml = L & 15;
    int kl = 16 * halfL + e;

    float val;
    if (region == 0) {                 // W_fuse (512,256): A[m][k] = Wf[k][m]
        int m = fi & 3, kt = (fi >> 2) & 15, q = fi >> 6;
        int mt = q * 4 + m;
        val = Wf[(kt * 32 + kl) * Hdim + (mt * 16 + ml)];
    } else if (region == 1) {          // W_a (256,256)
        int m = fi & 3, kt = (fi >> 2) & 7, q = fi >> 5;
        int mt = q * 4 + m;
        val = Wa[(kt * 32 + kl) * Hdim + (mt * 16 + ml)];
    } else {                           // [W_smax | W_smoid] (256,16)
        int k = fi * 32 + kl;
        val = (ml < 8) ? Wsx[k * Edim + ml] : Wsg[k * Edim + (ml - 8)];
    }
    ws[t] = (__bf16)val;
}

// ---------------------------------------------------------------------------
// Fused MoE-gate kernel. Block = 8 waves, 128 rows, 132KB LDS (K-split fuse).
// B-fragment: lane L: n=L&15 (row), half=L>>4, element e -> K = 16*half+e.
// C/D: element v -> m = v+8*half, lane -> n = L&15.
// Wave tile = 4 mtiles x 4 row-groups (C stationary, 128 acc VGPRs).
// ---------------------------------------------------------------------------
__global__ __launch_bounds__(256) void moe_fused_kernel(
    const float* __restrict__ s_e, const float* __restrict__ p_e,
    const float* __restrict__ b_fuse, const float* __restrict__ b_a,
    const float* __restrict__ b_smax, const float* __restrict__ b_smoid,
    const float* __restrict__ W_out, const float* __restrict__ b_out,
    const __bf16* __restrict__ ws,
    float* __restrict__ out_main, float* __restrict__ out_pmax,
    float* __restrict__ out_pmoid)
{
    extern __shared__ char smem[];
    char* sXbuf = smem;                 // staged input half (and later g)
    char* sE    = smem + BUFBYTES;      // e (bf16)
    char* sG    = smem;                 // g overlays Xbuf

    const int tid  = threadIdx.x;
    const int lane = tid & 31;
    const int n    = lane & 15;
    const int half = lane >> 4;
    const int w    = tid >> 5;
    const int row0 = blockIdx.x * ROWS_PB;
    const v16bf* wv = (const v16bf*)ws;

    const int q  = w & 3;               // mt quad (4 waves cover 16 mtiles)
    const int g0 = (w >> 2) * 4;        // 2 halves cover 8 row-groups
    const int mt0 = q * 4;

    v8f acc[4][4];
    {
        const v8f z = {0.f,0.f,0.f,0.f,0.f,0.f,0.f,0.f};
#pragma unroll
        for (int m = 0; m < 4; ++m)
#pragma unroll
            for (int g = 0; g < 4; ++g) acc[m][g] = z;
    }

    // ================= Fuse GEMM, K-split over {s_e, p_e} ==================
#pragma unroll
    for (int stage = 0; stage < 2; ++stage) {
        const float* src0 = stage ? p_e : s_e;
        // stage 128 rows x 256 feats f32 -> bf16 LDS
#pragma unroll
        for (int i = 0; i < 32; ++i) {
            int flat = tid + 256 * i;        // 128 rows x 64 float4 chunks
            int r = flat >> 6, c = flat & 63;
            float4 f = *(const float4*)(src0 + (size_t)(row0 + r) * Hdim + c * 4);
            uint2 d; d.x = pack_bf16(f.x, f.y); d.y = pack_bf16(f.z, f.w);
            *(uint2*)(sXbuf + r * SSTRIDE + c * 8) = d;
        }
        __syncthreads();

#pragma unroll
        for (int ktl = 0; ktl < 8; ++ktl) {
            const int kt = stage * 8 + ktl;         // weight K-tile 0..15
            // prefetch A-stream 2 K-steps ahead (4KB contiguous chunk)
            if (kt + 2 < 16)
                __builtin_prefetch((const char*)(wv + WF_BASE
                    + ((q * 16 + (kt + 2)) * 4) * 32) + lane * 32, 0, 3);
            v16bf A[4], Bv[4];
#pragma unroll
            for (int m = 0; m < 4; ++m)
                A[m] = wv[WF_BASE + ((q * 16 + kt) * 4 + m) * 32 + lane];
#pragma unroll
            for (int g = 0; g < 4; ++g)
                Bv[g] = load_frag16(sXbuf + ((g0 + g) * 16 + n) * SSTRIDE
                                          + (ktl * 32 + half * 16) * 2);
#pragma unroll
            for (int m = 0; m < 4; ++m)
#pragma unroll
                for (int g = 0; g < 4; ++g)
                    acc[m][g] = __builtin_amdgcn_wmma_f32_16x16x32_bf16(
                        false, A[m], false, Bv[g], (short)0, acc[m][g], false, false);
        }
        __syncthreads();   // before restaging / before g overlays
    }

    // fuse epilogue: bias + relu -> bf16 e tiles in LDS
#pragma unroll
    for (int m = 0; m < 4; ++m) {
        const float* bp = b_fuse + (mt0 + m) * 16 + half * 8;
        float bb[8];
#pragma unroll
        for (int v = 0; v < 8; ++v) bb[v] = bp[v];
#pragma unroll
        for (int g = 0; g < 4; ++g) {
            float r[8];
#pragma unroll
            for (int v = 0; v < 8; ++v) {
                float x = acc[m][g][v] + bb[v];
                r[v] = x > 0.f ? x : 0.f;
            }
            uint4 st = { pack_bf16(r[0], r[1]), pack_bf16(r[2], r[3]),
                         pack_bf16(r[4], r[5]), pack_bf16(r[6], r[7]) };
            *(uint4*)(sE + ((g0 + g) * 16 + n) * SSTRIDE
                         + (16 * (mt0 + m) + 8 * half) * 2) = st;
        }
    }
    __syncthreads();

    // ================= GEMM2: g^T = tanh(Wa^T e^T + b_a) ===================
    {
#pragma unroll
        for (int m = 0; m < 4; ++m)
#pragma unroll
            for (int g = 0; g < 4; ++g) {
                const v8f z = {0.f,0.f,0.f,0.f,0.f,0.f,0.f,0.f};
                acc[m][g] = z;
            }
#pragma unroll
        for (int kt = 0; kt < 8; ++kt) {
            if (kt + 2 < 8)
                __builtin_prefetch((const char*)(wv + WA_BASE
                    + ((q * 8 + (kt + 2)) * 4) * 32) + lane * 32, 0, 3);
            v16bf A[4], Bv[4];
#pragma unroll
            for (int m = 0; m < 4; ++m)
                A[m] = wv[WA_BASE + ((q * 8 + kt) * 4 + m) * 32 + lane];
#pragma unroll
            for (int g = 0; g < 4; ++g)
                Bv[g] = load_frag16(sE + ((g0 + g) * 16 + n) * SSTRIDE
                                       + (kt * 32 + half * 16) * 2);
#pragma unroll
            for (int m = 0; m < 4; ++m)
#pragma unroll
                for (int g = 0; g < 4; ++g)
                    acc[m][g] = __builtin_amdgcn_wmma_f32_16x16x32_bf16(
                        false, A[m], false, Bv[g], (short)0, acc[m][g], false, false);
        }
#pragma unroll
        for (int m = 0; m < 4; ++m) {
            const float* bp = b_a + (mt0 + m) * 16 + half * 8;
            float bb[8];
#pragma unroll
            for (int v = 0; v < 8; ++v) bb[v] = bp[v];
#pragma unroll
            for (int g = 0; g < 4; ++g) {
                float r[8];
#pragma unroll
                for (int v = 0; v < 8; ++v) r[v] = fast_tanh(acc[m][g][v] + bb[v]);
                uint4 st = { pack_bf16(r[0], r[1]), pack_bf16(r[2], r[3]),
                             pack_bf16(r[4], r[5]), pack_bf16(r[6], r[7]) };
                *(uint4*)(sG + ((g0 + g) * 16 + n) * SSTRIDE
                             + (16 * (mt0 + m) + 8 * half) * 2) = st;
            }
        }
    }
    __syncthreads();

    // ================= GEMM3: logits^T ; wave w owns row-group w ===========
    v8f acc3 = {0.f,0.f,0.f,0.f,0.f,0.f,0.f,0.f};
#pragma unroll
    for (int kt = 0; kt < 8; ++kt) {
        v16bf A  = wv[WG_BASE + kt * 32 + lane];
        v16bf Bv = load_frag16(sG + (16 * w + n) * SSTRIDE
                                  + (kt * 32 + half * 16) * 2);
        acc3 = __builtin_amdgcn_wmma_f32_16x16x32_bf16(
            false, A, false, Bv, (short)0, acc3, false, false);
    }

    // Epilogue: half0 lanes hold the 8 smax logits of row n, half1 the smoid.
    const float* bsel = half ? b_smoid : b_smax;
    float lg[8];
#pragma unroll
    for (int v = 0; v < 8; ++v) lg[v] = acc3[v] + bsel[v];

    float mx = lg[0]; int idx = 0;           // first-max == jnp.argmax
#pragma unroll
    for (int v = 1; v < 8; ++v) { if (lg[v] > mx) { mx = lg[v]; idx = v; } }

    float ex[8], s = 0.f;
#pragma unroll
    for (int v = 0; v < 8; ++v) { ex[v] = __expf(lg[v] - mx); s += ex[v]; }
    float inv = 1.f / s;

    const int grow = row0 + 16 * w + n;
    if (half == 0) {
        float4 o0 = {ex[0]*inv, ex[1]*inv, ex[2]*inv, ex[3]*inv};
        float4 o1 = {ex[4]*inv, ex[5]*inv, ex[6]*inv, ex[7]*inv};
        ((float4*)(out_pmax + (size_t)grow * 8))[0] = o0;
        ((float4*)(out_pmax + (size_t)grow * 8))[1] = o1;
    } else {
        float sg0 = 1.f/(1.f+__expf(-lg[0])), sg1 = 1.f/(1.f+__expf(-lg[1]));
        float sg2 = 1.f/(1.f+__expf(-lg[2])), sg3 = 1.f/(1.f+__expf(-lg[3]));
        float sg4 = 1.f/(1.f+__expf(-lg[4])), sg5 = 1.f/(1.f+__expf(-lg[5]));
        float sg6 = 1.f/(1.f+__expf(-lg[6])), sg7 = 1.f/(1.f+__expf(-lg[7]));
        float4 o0 = {sg0, sg1, sg2, sg3};
        float4 o1 = {sg4, sg5, sg6, sg7};
        ((float4*)(out_pmoid + (size_t)grow * 8))[0] = o0;
        ((float4*)(out_pmoid + (size_t)grow * 8))[1] = o1;
    }

    // out = dot(e[row, 32*idx .. +32], W_out[32*idx .. +32]) + b_out
    int idxb = __shfl(idx, n, 32);           // broadcast argmax from half0
    const char* ep = sE + (16 * w + n) * SSTRIDE + (32 * idxb + 16 * half) * 2;
    uint4 q0 = *(const uint4*)ep;
    uint4 q1 = *((const uint4*)ep + 1);
    const float* wo = W_out + 32 * idxb + 16 * half;
    float part = 0.f;
    part += bf_lo(q0.x) * wo[0]  + bf_hi(q0.x) * wo[1];
    part += bf_lo(q0.y) * wo[2]  + bf_hi(q0.y) * wo[3];
    part += bf_lo(q0.z) * wo[4]  + bf_hi(q0.z) * wo[5];
    part += bf_lo(q0.w) * wo[6]  + bf_hi(q0.w) * wo[7];
    part += bf_lo(q1.x) * wo[8]  + bf_hi(q1.x) * wo[9];
    part += bf_lo(q1.y) * wo[10] + bf_hi(q1.y) * wo[11];
    part += bf_lo(q1.z) * wo[12] + bf_hi(q1.z) * wo[13];
    part += bf_lo(q1.w) * wo[14] + bf_hi(q1.w) * wo[15];

    float part2 = __shfl_xor(part, 16, 32);
    if (half == 0) out_main[grow] = part + part2 + b_out[0];
}

// ---------------------------------------------------------------------------
extern "C" void kernel_launch(void* const* d_in, const int* in_sizes, int n_in,
                              void* d_out, int out_size, void* d_ws, size_t ws_size,
                              hipStream_t stream)
{
    const float* s_e     = (const float*)d_in[0];
    const float* p_e     = (const float*)d_in[1];
    const float* W_fuse  = (const float*)d_in[2];
    const float* b_fuse  = (const float*)d_in[3];
    const float* W_a     = (const float*)d_in[4];
    const float* b_a     = (const float*)d_in[5];
    const float* W_smax  = (const float*)d_in[6];
    const float* b_smax  = (const float*)d_in[7];
    const float* W_smoid = (const float*)d_in[8];
    const float* b_smoid = (const float*)d_in[9];
    const float* W_out   = (const float*)d_in[10];
    const float* b_out   = (const float*)d_in[11];

    const int Bn = in_sizes[0] / Hdim;                 // 262144
    __bf16* ws = (__bf16*)d_ws;
    float* out_main  = (float*)d_out;                  // [B]
    float* out_pmax  = out_main + (size_t)Bn;          // [B,8]
    float* out_pmoid = out_main + (size_t)Bn * 9;      // [B,8]

    prep_weights_kernel<<<(TOTAL_ELEMS + 255) / 256, 256, 0, stream>>>(
        W_fuse, W_a, W_smax, W_smoid, ws);

    const int blocks = Bn / ROWS_PB;                   // 2048
    moe_fused_kernel<<<blocks, 256, SMEM_BYTES, stream>>>(
        s_e, p_e, b_fuse, b_a, b_smax, b_smoid, W_out, b_out, ws,
        out_main, out_pmax, out_pmoid);
}
